// InterSelfAttention_S_84524956385360
// MI455X (gfx1250) — matrix operand order repaired and checked by reading, not compiled, not run on hardware
//
#include <hip/hip_runtime.h>
#include <hip/hip_bf16.h>

namespace {

constexpr int kDZ = 128;
constexpr int kH  = 4;
constexpr int kDH = 32;
constexpr int kDC = 128;
constexpr int kN  = 256;                     // tokens per (b,l) slice
constexpr float kScale = 0.17677669529663687f; // 1/sqrt(DH)

typedef __attribute__((ext_vector_type(16))) _Float16 v16h;
typedef __attribute__((ext_vector_type(8)))  float    v8f;

// 160 KB total: fits per-workgroup LDS whether the cap is 160 KB or 320 KB.
struct Smem {
  _Float16 z[kN * kDZ];          // 64 KB  layernormed activations (f16)
  _Float16 k[kN * kDH];          // 16 KB  per-head K
  _Float16 v[kN * kDH];          // 16 KB  per-head V
  _Float16 strip[8 * 16 * kN];   // 64 KB  per-wave 16-row score/staging strip
};

__device__ __forceinline__ v8f wmma16(v16h a, v16h b, v8f c) {
  return __builtin_amdgcn_wmma_f32_16x16x32_f16(false, a, false, b, (short)0, c,
                                                false, false);
}

// A fragment: 16x32 f16 tile at (row0,k0) of row-major LDS matrix, stride ld.
// Lanes 0-15: row m, K = k0+[0..15];
// lanes 16-31: same rows, K = k0+[16..31].
__device__ __forceinline__ v16h load_a(const _Float16* s, int row0, int k0,
                                       int ld, int lane) {
  const _Float16* p = s + (row0 + (lane & 15)) * ld + k0 + ((lane >> 4) << 4);
  v16h a;
#pragma unroll
  for (int e = 0; e < 16; ++e) a[e] = p[e];
  return a;
}

// B fragment: 32x16 f16 tile at (k0,n0), row-major f16 LDS, stride ld.
// VGPR v holds K=2v,2v+1 (lanes 0-15) / K=16+2v,16+2v+1 (lanes 16-31); N=lane&15.
__device__ __forceinline__ v16h load_b(const _Float16* s, int k0, int n0,
                                       int ld, int lane) {
  const _Float16* p = s + (k0 + ((lane >> 4) << 4)) * ld + n0 + (lane & 15);
  v16h b;
#pragma unroll
  for (int e = 0; e < 16; ++e) b[e] = p[e * ld];
  return b;
}

// B fragment from row-major f32 global weight matrix, converted to f16.
__device__ __forceinline__ v16h load_b_gf32(const float* w, int k0, int n0,
                                            int ld, int lane) {
  const float* p = w + (k0 + ((lane >> 4) << 4)) * ld + n0 + (lane & 15);
  v16h b;
#pragma unroll
  for (int e = 0; e < 16; ++e) b[e] = (_Float16)p[e * ld];
  return b;
}

// B fragment of a TRANSPOSED LDS matrix: B[k][n] = s[n][k]; contiguous per lane.
__device__ __forceinline__ v16h load_b_tr(const _Float16* s, int k0, int n0,
                                          int ld, int lane) {
  const _Float16* p = s + (n0 + (lane & 15)) * ld + k0 + ((lane >> 4) << 4);
  v16h b;
#pragma unroll
  for (int e = 0; e < 16; ++e) b[e] = p[e];
  return b;
}

__device__ __forceinline__ float wave_sum(float x) {
#pragma unroll
  for (int off = 16; off > 0; off >>= 1) x += __shfl_xor(x, off, 32);
  return x;
}
__device__ __forceinline__ float wave_max(float x) {
#pragma unroll
  for (int off = 16; off > 0; off >>= 1) x = fmaxf(x, __shfl_xor(x, off, 32));
  return x;
}

__global__ __launch_bounds__(256)
void inter_self_attention_kernel(
    const float* __restrict__ z_inter, const float* __restrict__ dist,
    const float* __restrict__ ln_g, const float* __restrict__ ln_b,
    const float* __restrict__ Wq, const float* __restrict__ bq,
    const float* __restrict__ Wk, const float* __restrict__ bk,
    const float* __restrict__ Wv, const float* __restrict__ bv,
    const float* __restrict__ Wg, const float* __restrict__ bg,
    const float* __restrict__ Wo, const float* __restrict__ bo,
    float* __restrict__ out) {
  __shared__ Smem sm;
  const int l    = blockIdx.x;
  const int lane = threadIdx.x & 31;
  const int wave = threadIdx.x >> 5;

  const float* zl = z_inter + (size_t)l * kN * kDZ;
  const float* dl = dist    + (size_t)l * kN * kN;
  float*       ol = out     + (size_t)l * kN * kDZ;

  // ---------------- LayerNorm: one wave per token row ----------------
  for (int i = wave; i < kN; i += 8) {
    const float* p = zl + i * kDZ;
    float x0 = p[lane], x1 = p[lane + 32], x2 = p[lane + 64], x3 = p[lane + 96];
    float mu = wave_sum(x0 + x1 + x2 + x3) * (1.0f / kDZ);
    float d0 = x0 - mu, d1 = x1 - mu, d2 = x2 - mu, d3 = x3 - mu;
    float var = wave_sum(d0 * d0 + d1 * d1 + d2 * d2 + d3 * d3) * (1.0f / kDZ);
    float rstd = rsqrtf(var + 1e-5f);
    sm.z[i * kDZ + lane     ] = (_Float16)(d0 * rstd * ln_g[lane     ] + ln_b[lane     ]);
    sm.z[i * kDZ + lane + 32] = (_Float16)(d1 * rstd * ln_g[lane + 32] + ln_b[lane + 32]);
    sm.z[i * kDZ + lane + 64] = (_Float16)(d2 * rstd * ln_g[lane + 64] + ln_b[lane + 64]);
    sm.z[i * kDZ + lane + 96] = (_Float16)(d3 * rstd * ln_g[lane + 96] + ln_b[lane + 96]);
  }
  __syncthreads();

  const v8f vzero = {};

#pragma unroll 1
  for (int pass = 0; pass < 2; ++pass) {
    const int mOwn = (pass * 8 + wave) * 16;  // this wave's 16-row block
    _Float16* strip = sm.strip + wave * (16 * kN);

    v8f oacc[8];
#pragma unroll
    for (int i = 0; i < 8; ++i) oacc[i] = vzero;

#pragma unroll 1
    for (int h = 0; h < kH; ++h) {
      const int hc = h * kDH;

      // ---- cooperative K,V projections for all 256 rows ----
#pragma unroll
      for (int nt = 0; nt < 2; ++nt) {
        const int n0 = nt * 16;
        v16h bkf[4], bvf[4];
#pragma unroll
        for (int kc = 0; kc < 4; ++kc) {
          bkf[kc] = load_b_gf32(Wk, kc * 32, hc + n0, kDC, lane);
          bvf[kc] = load_b_gf32(Wv, kc * 32, hc + n0, kDC, lane);
        }
#pragma unroll
        for (int mt = 0; mt < 2; ++mt) {
          const int m0 = (wave * 2 + mt) * 16;
          v8f ak = vzero, av = vzero;
#pragma unroll
          for (int kc = 0; kc < 4; ++kc) {
            v16h a = load_a(sm.z, m0, kc * 32, kDZ, lane);
            ak = wmma16(a, bkf[kc], ak);
            av = wmma16(a, bvf[kc], av);
          }
          const int n = n0 + (lane & 15);
          const float bkv = bk[hc + n], bvv = bv[hc + n];
#pragma unroll
          for (int r = 0; r < 8; ++r) {
            const int row = m0 + r + ((lane >> 4) << 3);
            sm.k[row * kDH + n] = (_Float16)(ak[r] + bkv);
            sm.v[row * kDH + n] = (_Float16)(av[r] + bvv);
          }
        }
      }
      __syncthreads();

      // ---- Q tile for own rows (scaled, biased), staged to A layout ----
#pragma unroll
      for (int nt = 0; nt < 2; ++nt) {
        const int n0 = nt * 16;
        v8f aq = vzero;
#pragma unroll
        for (int kc = 0; kc < 4; ++kc) {
          v16h a = load_a(sm.z, mOwn, kc * 32, kDZ, lane);
          v16h b = load_b_gf32(Wq, kc * 32, hc + n0, kDC, lane);
          aq = wmma16(a, b, aq);
        }
        const int n = n0 + (lane & 15);
        const float bqv = bq[hc + n];
#pragma unroll
        for (int r = 0; r < 8; ++r)
          strip[(r + ((lane >> 4) << 3)) * kDH + n] =
              (_Float16)((aq[r] + bqv) * kScale);
      }
      // same-wave LDS ordering: store->load in order per the DS pipeline
      v16h qfrag = load_a(strip, 0, 0, kDH, lane);

      // ---- S = q . K^T  (K-dim = 32: one wmma per 16x16 tile) ----
#pragma unroll 1
      for (int ni = 0; ni < 16; ++ni) {
        v16h b = load_b_tr(sm.k, 0, ni * 16, kDH, lane);
        v8f acc = wmma16(qfrag, b, vzero);
        const int col = ni * 16 + (lane & 15);
#pragma unroll
        for (int r = 0; r < 8; ++r)
          strip[(r + ((lane >> 4) << 3)) * kN + col] = (_Float16)acc[r];
      }

      // ---- distance coefficient + softmax over j (whole wave per row) ----
#pragma unroll 1
      for (int i = 0; i < 16; ++i) {
        const int grow = mOwn + i;
        if (grow + 1 < kN)
          __builtin_prefetch(dl + (grow + 1) * kN + lane * 8, 0, 0);
        float vals[8];
        float mx = -3.402823466e38f;
#pragma unroll
        for (int jt = 0; jt < 8; ++jt) {
          const int j = jt * 32 + lane;
          const float d = dl[grow * kN + j] * 0.125f;
          const float coef = __expf(-0.5f * d * d);
          const float sc = (float)strip[i * kN + j] * coef;
          vals[jt] = sc;
          mx = fmaxf(mx, sc);
        }
        mx = wave_max(mx);
        float sum = 0.f;
#pragma unroll
        for (int jt = 0; jt < 8; ++jt) {
          vals[jt] = __expf(vals[jt] - mx);
          sum += vals[jt];
        }
        sum = wave_sum(sum);
        const float inv = 1.0f / sum;
#pragma unroll
        for (int jt = 0; jt < 8; ++jt)
          strip[i * kN + jt * 32 + lane] = (_Float16)(vals[jt] * inv);
      }

      // ---- O = P . V for own 16 rows ----
      v8f accO[2] = {};
#pragma unroll 1
      for (int kc = 0; kc < 8; ++kc) {
        v16h a = load_a(strip, 0, kc * 32, kN, lane);
#pragma unroll
        for (int nt = 0; nt < 2; ++nt) {
          v16h b = load_b(sm.v, kc * 32, nt * 16, kDH, lane);
          accO[nt] = wmma16(a, b, accO[nt]);
        }
      }

      // ---- gate = sigmoid(z.Wg + bg); stage X = O*gate to A layout ----
#pragma unroll
      for (int nt = 0; nt < 2; ++nt) {
        v8f accG = vzero;
#pragma unroll
        for (int kc = 0; kc < 4; ++kc) {
          v16h a = load_a(sm.z, mOwn, kc * 32, kDZ, lane);
          v16h b = load_b_gf32(Wg, kc * 32, hc + nt * 16, kDC, lane);
          accG = wmma16(a, b, accG);
        }
        const int n = nt * 16 + (lane & 15);
        const float bgv = bg[hc + n];
#pragma unroll
        for (int r = 0; r < 8; ++r) {
          const float g = 1.0f / (1.0f + __expf(-(accG[r] + bgv)));
          strip[(r + ((lane >> 4) << 3)) * kDH + n] =
              (_Float16)(accO[nt][r] * g);
        }
      }
      v16h xfrag = load_a(strip, 0, 0, kDH, lane);

      // ---- fused output projection: oacc += X . Wo[hc:hc+32, :] ----
#pragma unroll 1
      for (int nt = 0; nt < 8; ++nt) {
        v16h b = load_b_gf32(Wo, hc, nt * 16, kDZ, lane);
        oacc[nt] = wmma16(xfrag, b, oacc[nt]);
      }

      __syncthreads();  // protect sm.k / sm.v before next head rewrites them
    }  // heads

    // ---- epilogue: write this wave's 16 rows ----
#pragma unroll 1
    for (int nt = 0; nt < 8; ++nt) {
      const int n = nt * 16 + (lane & 15);
      const float bov = bo[n];
#pragma unroll
      for (int r = 0; r < 8; ++r) {
        const int row = mOwn + r + ((lane >> 4) << 3);
        ol[row * kDZ + n] = oacc[nt][r] + bov;
      }
    }
  }  // passes
}

}  // namespace

extern "C" void kernel_launch(void* const* d_in, const int* in_sizes, int n_in,
                              void* d_out, int out_size, void* d_ws,
                              size_t ws_size, hipStream_t stream) {
  const float* z_inter = (const float*)d_in[0];
  const float* dist    = (const float*)d_in[1];
  const float* ln_g    = (const float*)d_in[2];
  const float* ln_b    = (const float*)d_in[3];
  const float* Wq = (const float*)d_in[4];  const float* bq = (const float*)d_in[5];
  const float* Wk = (const float*)d_in[6];  const float* bk = (const float*)d_in[7];
  const float* Wv = (const float*)d_in[8];  const float* bv = (const float*)d_in[9];
  const float* Wg = (const float*)d_in[10]; const float* bg = (const float*)d_in[11];
  const float* Wo = (const float*)d_in[12]; const float* bo = (const float*)d_in[13];
  float* out = (float*)d_out;
  (void)in_sizes; (void)n_in; (void)out_size; (void)d_ws; (void)ws_size;
  inter_self_attention_kernel<<<dim3(256), dim3(256), 0, stream>>>(
      z_inter, dist, ln_g, ln_b, Wq, bq, Wk, bk, Wv, bv, Wg, bg, Wo, bo, out);
}
